// KGCompletionGNN_43731357008329
// MI455X (gfx1250) — matrix operand order repaired and checked by reading, not compiled, not run on hardware
//
#include <hip/hip_runtime.h>

#define DEV __device__ __forceinline__

typedef __bf16 bf16_t;
typedef __attribute__((ext_vector_type(16))) bf16_t          v16bf;
typedef __attribute__((ext_vector_type(8)))  float           v8f;
typedef __attribute__((ext_vector_type(16))) unsigned short  v16u;
typedef __attribute__((ext_vector_type(8)))  unsigned short  v8u;

constexpr int N_  = 50000;
constexpr int M_  = 250000;
constexpr int D_  = 128;
constexpr int F_  = 768;
constexpr int R_  = 474;
constexpr int NQ_ = 1024;

// ---------- helpers ----------

DEV unsigned short f2b(float f) {                 // f32 -> bf16 (RNE)
  unsigned int u = __builtin_bit_cast(unsigned int, f);
  u += 0x7fffu + ((u >> 16) & 1u);
  return (unsigned short)(u >> 16);
}

DEV float leaky(float x) { return x >= 0.f ? x : 0.01f * x; }

DEV float red16(float v) {                        // sum over aligned 16-lane group
  v += __shfl_xor(v, 1, 32);
  v += __shfl_xor(v, 2, 32);
  v += __shfl_xor(v, 4, 32);
  v += __shfl_xor(v, 8, 32);
  return v;
}
DEV float red32(float v) { v = red16(v); v += __shfl_xor(v, 16, 32); return v; }

// One wave computes a 16x16 f32 tile of  A(16 x K) @ W^T  with W bf16 [128][kpad].
// A-fragment: lane l, row m=l&15; K = kk*32 + (l>>4)*16 + i (contig, matches B).
// B-fragment: col n=l&15, K = kk*32 + (l>>4)*16 + i (contig).
DEV v8f tile_gemm(const unsigned short* aL, const unsigned short* Wb,
                  int kpad, int kchunks, int dbase, v8f acc) {
  const int lane = threadIdx.x & 31;
  const int hl   = lane >> 4;
  const int mn   = lane & 15;
  const unsigned short* arow = aL + mn * kpad + hl * 16;
  const unsigned short* brow = Wb + (dbase + mn) * kpad + hl * 16;
  for (int kk = 0; kk < kchunks; ++kk) {
    v8u  a0 = *(const v8u*)(arow + kk * 32);
    v8u  a1 = *(const v8u*)(arow + kk * 32 + 8);
    v16u bv = *(const v16u*)(brow + kk * 32);
    v16u av;
#pragma unroll
    for (int i = 0; i < 8; ++i) { av[i] = a0[i]; av[i + 8] = a1[i]; }
    acc = __builtin_amdgcn_wmma_f32_16x16x32_bf16(
            false, __builtin_bit_cast(v16bf, av),
            false, __builtin_bit_cast(v16bf, bv),
            (short)0, acc, false, false);
  }
  return acc;
}

DEV v8f zero8() { v8f a; 
#pragma unroll
  for (int i = 0; i < 8; ++i) a[i] = 0.f; 
  return a; }

// ---------- utility kernels ----------

__global__ void k_cvt(const float* __restrict__ src, unsigned short* __restrict__ dst,
                      int rows, int kreal, int kpad) {
  int i = blockIdx.x * 256 + threadIdx.x;
  if (i < rows * kpad) {
    int r = i / kpad, k = i % kpad;
    dst[i] = f2b(k < kreal ? src[r * kreal + k] : 0.f);
  }
}

__global__ void k_fill(float* __restrict__ p, float v, int n) {
  int i = blockIdx.x * 256 + threadIdx.x;
  if (i < n) p[i] = v;
}

__global__ void k_count(const int* __restrict__ ht, float* __restrict__ cnt, int m) {
  int i = blockIdx.x * 256 + threadIdx.x;
  if (i < m) {
    atomicAdd(&cnt[ht[2 * i + 1]], 1.f);   // forward messages land on tails
    atomicAdd(&cnt[ht[2 * i]], 1.f);       // backward messages land on heads
  }
}

// ---------- H0 = LN(leaky(entity_feat @ W_nin^T + b)) ----------

__global__ void k_node_in(const float* __restrict__ ent, const unsigned short* __restrict__ Wb,
                          const float* __restrict__ bias, float* __restrict__ H0,
                          float* __restrict__ H) {
  __shared__ __attribute__((aligned(16))) unsigned short aL[16 * 768];
  __shared__ float oL[16 * 128];
  const int tid = threadIdx.x, tile = blockIdx.x;
  for (int idx = tid; idx < 16 * 768; idx += 256) {
    int e = idx / 768, k = idx % 768;
    aL[idx] = f2b(ent[(tile * 16 + e) * 768 + k]);
  }
  __syncthreads();
  const int wave = tid >> 5, lane = tid & 31, mn = lane & 15, hl = lane >> 4;
  const int dbase = wave * 16;
  v8f acc = zero8();
  acc = tile_gemm(aL, Wb, 768, 24, dbase, acc);
  float bd = bias[dbase + mn];
#pragma unroll
  for (int v = 0; v < 8; ++v) {
    int m = v + hl * 8;
    oL[m * 128 + dbase + mn] = leaky(acc[v] + bd);
  }
  __syncthreads();
  // LayerNorm: wave handles rows 2w (lanes 0-15) and 2w+1 (lanes 16-31)
  int row = 2 * wave + hl;
  float vals[8]; float s = 0.f;
#pragma unroll
  for (int i = 0; i < 8; ++i) { vals[i] = oL[row * 128 + mn * 8 + i]; s += vals[i]; }
  float mean = red16(s) * (1.f / 128.f);
  float vs = 0.f;
#pragma unroll
  for (int i = 0; i < 8; ++i) { float d = vals[i] - mean; vs += d * d; }
  float inv = rsqrtf(red16(vs) * (1.f / 128.f) + 1e-5f);
  int grow = tile * 16 + row;
#pragma unroll
  for (int i = 0; i < 8; ++i) {
    float o = (vals[i] - mean) * inv;
    H0[grow * 128 + mn * 8 + i] = o;
    H[grow * 128 + mn * 8 + i] = o;
  }
}

// ---------- E0 combos: 948 distinct (relation, query) rows ----------

__global__ void k_edge_in(const float* __restrict__ relf, const unsigned short* __restrict__ Wb,
                          const float* __restrict__ bias, float* __restrict__ ecomb) {
  __shared__ __attribute__((aligned(16))) unsigned short aL[16 * 800];
  __shared__ float oL[16 * 128];
  const int tid = threadIdx.x, tile = blockIdx.x;
  for (int idx = tid; idx < 16 * 800; idx += 256) {
    int e = idx / 800, k = idx % 800;
    int c = tile * 16 + e;
    float v = 0.f;
    if (c < 2 * R_) {
      int r = c % R_, q = c / R_;
      if (k < 768) v = relf[r * 768 + k];
      else if (k == 768) v = (float)q;
    }
    aL[idx] = f2b(v);
  }
  __syncthreads();
  const int wave = tid >> 5, lane = tid & 31, mn = lane & 15, hl = lane >> 4;
  const int dbase = wave * 16;
  v8f acc = zero8();
  acc = tile_gemm(aL, Wb, 800, 25, dbase, acc);
  float bd = bias[dbase + mn];
#pragma unroll
  for (int v = 0; v < 8; ++v) {
    int m = v + hl * 8;
    oL[m * 128 + dbase + mn] = leaky(acc[v] + bd);
  }
  __syncthreads();
  int row = 2 * wave + hl;
  float vals[8]; float s = 0.f;
#pragma unroll
  for (int i = 0; i < 8; ++i) { vals[i] = oL[row * 128 + mn * 8 + i]; s += vals[i]; }
  float mean = red16(s) * (1.f / 128.f);
  float vs = 0.f;
#pragma unroll
  for (int i = 0; i < 8; ++i) { float d = vals[i] - mean; vs += d * d; }
  float inv = rsqrtf(red16(vs) * (1.f / 128.f) + 1e-5f);
  int c = tile * 16 + row;
  if (c < 2 * R_) {
#pragma unroll
    for (int i = 0; i < 8; ++i)
      ecomb[c * 128 + mn * 8 + i] = (vals[i] - mean) * inv;
  }
}

__global__ void k_scatterE(const int* __restrict__ r_t, const float* __restrict__ ecomb,
                           float* __restrict__ E) {
  int i = blockIdx.x * 256 + threadIdx.x;
  if (i < M_ * D_) {
    int m = i >> 7, d = i & 127;
    int c = r_t[m] + (m < NQ_ ? R_ : 0);
    E[i] = ecomb[c * 128 + d];
  }
}

// ---------- message pass: msg = ([H_src,E,H_src*r,E*r] @ W^T + b)*p -> atomic agg[dst] ----------

__global__ void k_msg(const int* __restrict__ ht, const int* __restrict__ r_t,
                      const float* __restrict__ psel, const float* __restrict__ H,
                      const float* __restrict__ E, const float* __restrict__ rel_emb,
                      const unsigned short* __restrict__ Wf, const float* __restrict__ bf,
                      const unsigned short* __restrict__ Wbk, const float* __restrict__ bb,
                      float* __restrict__ agg) {
  __shared__ __attribute__((aligned(16))) unsigned short aL[16 * 512];
  __shared__ int srcL[16], dstL[16], rL[16];
  __shared__ float pL[16];
  const int tid = threadIdx.x, tile = blockIdx.x, dir = blockIdx.y;
  if (tid < 16) {
    int m = tile * 16 + tid;
    int h = ht[2 * m], t = ht[2 * m + 1];
    srcL[tid] = dir ? t : h;     // forward reads H[head], backward reads H[tail]
    dstL[tid] = dir ? h : t;     // forward scatters to tail, backward to head
    rL[tid] = r_t[m];
    float ps = psel[m];
    pL[tid] = ps / ps;
  }
  __syncthreads();
  for (int idx = tid; idx < 16 * 512; idx += 256) {
    int e = idx >> 9, k = idx & 511;
    int m = tile * 16 + e;
    int s = srcL[e], r = rL[e], kk = k & 127;
    float v;
    if (k < 128)      v = H[s * 128 + kk];
    else if (k < 256) v = E[m * 128 + kk];
    else if (k < 384) v = H[s * 128 + kk] * rel_emb[r * 128 + kk];
    else              v = E[m * 128 + kk] * rel_emb[r * 128 + kk];
    aL[idx] = f2b(v);
  }
  __syncthreads();
  const unsigned short* W = dir ? Wbk : Wf;
  const float* bias = dir ? bb : bf;
  const int wave = tid >> 5, lane = tid & 31, mn = lane & 15, hl = lane >> 4;
  const int dbase = wave * 16;
  v8f acc = zero8();
  acc = tile_gemm(aL, W, 512, 16, dbase, acc);
  float bd = bias[dbase + mn];
#pragma unroll
  for (int v = 0; v < 8; ++v) {
    int m = v + hl * 8;
    float val = (acc[v] + bd) * pL[m];
    atomicAdd(&agg[dstL[m] * 128 + dbase + mn], val);
  }
}

// ---------- H = LN(leaky(agg/max(cnt,1)) + H) ----------

__global__ void k_node_upd(const float* __restrict__ agg, const float* __restrict__ cnt,
                           float* __restrict__ H) {
  const int wave = threadIdx.x >> 5, lane = threadIdx.x & 31;
  const int row = blockIdx.x * 8 + wave;
  float invc = 1.f / fmaxf(cnt[row], 1.f);
  float vals[4]; float s = 0.f;
#pragma unroll
  for (int i = 0; i < 4; ++i) {
    int c = lane * 4 + i;
    float a = leaky(agg[row * 128 + c] * invc);
    vals[i] = a + H[row * 128 + c];
    s += vals[i];
  }
  float mean = red32(s) * (1.f / 128.f);
  float vs = 0.f;
#pragma unroll
  for (int i = 0; i < 4; ++i) { float d = vals[i] - mean; vs += d * d; }
  float inv = rsqrtf(red32(vs) * (1.f / 128.f) + 1e-5f);
#pragma unroll
  for (int i = 0; i < 4; ++i)
    H[row * 128 + lane * 4 + i] = (vals[i] - mean) * inv;
}

// ---------- E = LN(leaky([H_h,E,H_t] @ eu_W^T + b) + E) ----------

__global__ void k_edge_upd(const int* __restrict__ ht, const float* __restrict__ H,
                           float* __restrict__ E, const unsigned short* __restrict__ W,
                           const float* __restrict__ bias) {
  __shared__ __attribute__((aligned(16))) unsigned short aL[16 * 384];
  __shared__ float oL[16 * 128];
  __shared__ int hL[16], tL[16];
  const int tid = threadIdx.x, tile = blockIdx.x;
  if (tid < 16) {
    int m = tile * 16 + tid;
    hL[tid] = ht[2 * m]; tL[tid] = ht[2 * m + 1];
  }
  __syncthreads();
  for (int idx = tid; idx < 16 * 384; idx += 256) {
    int e = idx / 384, k = idx % 384;
    int m = tile * 16 + e, kk = k & 127;
    float v;
    if (k < 128)      v = H[hL[e] * 128 + kk];
    else if (k < 256) v = E[m * 128 + kk];
    else              v = H[tL[e] * 128 + kk];
    aL[idx] = f2b(v);
  }
  __syncthreads();
  const int wave = tid >> 5, lane = tid & 31, mn = lane & 15, hl = lane >> 4;
  const int dbase = wave * 16;
  v8f acc = zero8();
  acc = tile_gemm(aL, W, 384, 12, dbase, acc);
  float bd = bias[dbase + mn];
#pragma unroll
  for (int v = 0; v < 8; ++v) {
    int m = v + hl * 8;
    oL[m * 128 + dbase + mn] = leaky(acc[v] + bd);
  }
  __syncthreads();
  int row = 2 * wave + hl;
  int m = tile * 16 + row;
  float vals[8]; float s = 0.f;
#pragma unroll
  for (int i = 0; i < 8; ++i) {
    vals[i] = oL[row * 128 + mn * 8 + i] + E[m * 128 + mn * 8 + i];  // residual (fp32)
    s += vals[i];
  }
  float mean = red16(s) * (1.f / 128.f);
  float vs = 0.f;
#pragma unroll
  for (int i = 0; i < 8; ++i) { float d = vals[i] - mean; vs += d * d; }
  float inv = rsqrtf(red16(vs) * (1.f / 128.f) + 1e-5f);
#pragma unroll
  for (int i = 0; i < 8; ++i)
    E[m * 128 + mn * 8 + i] = (vals[i] - mean) * inv;
}

// ---------- classifier over the first NQ edges ----------

__global__ void k_cls(const int* __restrict__ ht, const int* __restrict__ r_t,
                      const float* __restrict__ E, const float* __restrict__ ecomb,
                      const float* __restrict__ H, const float* __restrict__ H0,
                      const unsigned short* __restrict__ W1b, const float* __restrict__ b1,
                      const float* __restrict__ W2, const float* __restrict__ b2,
                      float* __restrict__ out) {
  __shared__ __attribute__((aligned(16))) unsigned short aL[16 * 768];
  __shared__ float oL[16 * 128];
  __shared__ int hL[16], tL[16], cL[16];
  const int tid = threadIdx.x, tile = blockIdx.x;
  if (tid < 16) {
    int m = tile * 16 + tid;               // qidx == arange(NQ)
    hL[tid] = ht[2 * m]; tL[tid] = ht[2 * m + 1];
    cL[tid] = r_t[m] + R_;                 // E0 for query edges (q == 1)
  }
  __syncthreads();
  for (int idx = tid; idx < 16 * 768; idx += 256) {
    int e = idx / 768, k = idx % 768;
    int m = tile * 16 + e, kk = k & 127;
    float v;
    switch (k >> 7) {
      case 0:  v = E[m * 128 + kk];         break;
      case 1:  v = ecomb[cL[e] * 128 + kk]; break;
      case 2:  v = H[hL[e] * 128 + kk];     break;
      case 3:  v = H0[hL[e] * 128 + kk];    break;
      case 4:  v = H[tL[e] * 128 + kk];     break;
      default: v = H0[tL[e] * 128 + kk];    break;
    }
    aL[idx] = f2b(v);
  }
  __syncthreads();
  const int wave = tid >> 5, lane = tid & 31, mn = lane & 15, hl = lane >> 4;
  const int dbase = wave * 16;
  v8f acc = zero8();
  acc = tile_gemm(aL, W1b, 768, 24, dbase, acc);
  float bd = b1[dbase + mn];
#pragma unroll
  for (int v = 0; v < 8; ++v) {
    int m = v + hl * 8;
    oL[m * 128 + dbase + mn] = leaky(acc[v] + bd);
  }
  __syncthreads();
  int row = 2 * wave + hl;
  float s = 0.f;
#pragma unroll
  for (int i = 0; i < 8; ++i)
    s += oL[row * 128 + mn * 8 + i] * W2[mn * 8 + i];
  s = red16(s);
  if (mn == 0) out[tile * 16 + row] = s + b2[0];
}

// ---------- launch ----------

extern "C" void kernel_launch(void* const* d_in, const int* in_sizes, int n_in,
                              void* d_out, int out_size, void* d_ws, size_t ws_size,
                              hipStream_t stream) {
  const int*   ht      = (const int*)  d_in[0];
  const int*   r_t     = (const int*)  d_in[1];
  const float* entf    = (const float*)d_in[2];
  const float* relf    = (const float*)d_in[3];
  const float* psel    = (const float*)d_in[4];
  // d_in[5] (queries) implied by m < NQ
  const float* rel_emb = (const float*)d_in[6];
  const float* W_ein   = (const float*)d_in[7];
  const float* b_ein   = (const float*)d_in[8];
  const float* W_nin   = (const float*)d_in[9];
  const float* b_nin   = (const float*)d_in[10];
  const float* mpf_W   = (const float*)d_in[11];
  const float* mpf_b   = (const float*)d_in[12];
  const float* mpb_W   = (const float*)d_in[13];
  const float* mpb_b   = (const float*)d_in[14];
  const float* eu_W    = (const float*)d_in[15];
  const float* eu_b    = (const float*)d_in[16];
  const float* cls1_W  = (const float*)d_in[17];
  const float* cls1_b  = (const float*)d_in[18];
  const float* cls2_W  = (const float*)d_in[19];
  const float* cls2_b  = (const float*)d_in[20];
  float* out = (float*)d_out;

  char* ws = (char*)d_ws;
  size_t off = 0;
  auto take = [&](size_t bytes) -> char* {
    char* p = ws + off;
    off = (off + bytes + 255) & ~(size_t)255;
    return p;
  };
  unsigned short* WninB = (unsigned short*)take((size_t)128 * 768 * 2);
  unsigned short* WeinB = (unsigned short*)take((size_t)128 * 800 * 2);
  unsigned short* mpfB  = (unsigned short*)take((size_t)2 * 128 * 512 * 2);
  unsigned short* mpbB  = (unsigned short*)take((size_t)2 * 128 * 512 * 2);
  unsigned short* euB   = (unsigned short*)take((size_t)2 * 128 * 384 * 2);
  unsigned short* c1B   = (unsigned short*)take((size_t)128 * 768 * 2);
  float* H0    = (float*)take((size_t)N_ * D_ * 4);
  float* H     = (float*)take((size_t)N_ * D_ * 4);
  float* E     = (float*)take((size_t)M_ * D_ * 4);
  float* ecomb = (float*)take((size_t)960 * D_ * 4);
  float* agg   = (float*)take((size_t)N_ * D_ * 4);
  float* cnt   = (float*)take((size_t)N_ * 4);

  auto nb = [](int n) { return (n + 255) / 256; };

  // weights -> bf16 (padded K where needed)
  k_cvt<<<nb(128 * 768), 256, 0, stream>>>(W_nin, WninB, 128, 768, 768);
  k_cvt<<<nb(128 * 800), 256, 0, stream>>>(W_ein, WeinB, 128, 769, 800);
  k_cvt<<<nb(256 * 512), 256, 0, stream>>>(mpf_W, mpfB, 256, 512, 512);
  k_cvt<<<nb(256 * 512), 256, 0, stream>>>(mpb_W, mpbB, 256, 512, 512);
  k_cvt<<<nb(256 * 384), 256, 0, stream>>>(eu_W, euB, 256, 384, 384);
  k_cvt<<<nb(128 * 768), 256, 0, stream>>>(cls1_W, c1B, 128, 768, 768);

  // degree counts (same every layer)
  k_fill<<<nb(N_), 256, 0, stream>>>(cnt, 0.f, N_);
  k_count<<<nb(M_), 256, 0, stream>>>(ht, cnt, M_);

  // input embeddings
  k_node_in<<<N_ / 16, 256, 0, stream>>>(entf, WninB, b_nin, H0, H);
  k_edge_in<<<60, 256, 0, stream>>>(relf, WeinB, b_ein, ecomb);
  k_scatterE<<<nb(M_ * D_), 256, 0, stream>>>(r_t, ecomb, E);

  // message-passing layers
  for (int l = 0; l < 2; ++l) {
    k_fill<<<nb(N_ * D_), 256, 0, stream>>>(agg, 0.f, N_ * D_);
    dim3 g(M_ / 16, 2);
    k_msg<<<g, 256, 0, stream>>>(ht, r_t, psel, H, E, rel_emb,
                                 mpfB + (size_t)l * 128 * 512, mpf_b + l * 128,
                                 mpbB + (size_t)l * 128 * 512, mpb_b + l * 128, agg);
    k_node_upd<<<N_ / 8, 256, 0, stream>>>(agg, cnt, H);
    k_edge_upd<<<M_ / 16, 256, 0, stream>>>(ht, H, E,
                                            euB + (size_t)l * 128 * 384, eu_b + l * 128);
  }

  // classifier
  k_cls<<<NQ_ / 16, 256, 0, stream>>>(ht, r_t, E, ecomb, H, H0,
                                      c1B, cls1_b, cls2_W, cls2_b, out);

  (void)in_sizes; (void)n_in; (void)out_size; (void)ws_size;
}
